// VectorQuantizer_67353677136258
// MI455X (gfx1250) — matrix operand order repaired and checked by reading, not compiled, not run on hardware
//
#include <hip/hip_runtime.h>

typedef __attribute__((ext_vector_type(16))) __bf16 v16bf;
typedef __attribute__((ext_vector_type(8)))  float  v8f;

#define D       256
#define HW      4096     // 64*64
#define CHW     1048576  // 256*4096
#define NPIX    65536    // 16*4096
#define NELEM   16777216 // 16*256*4096

// ws layout:
//   [0,      4096)   enorm  float[1024]
//   [4096,  528384)  emb_bf bf16[1024*256] (B-fragment swizzled)
//   [528384,790528)  idx    int[65536]

// ---- Kernel 1: codebook norms + BF16 swizzle into WMMA B-fragment order ----
__global__ void __launch_bounds__(256) vq_prep(const float* __restrict__ emb,
                                               float* __restrict__ enorm,
                                               __bf16* __restrict__ embb) {
  int row = blockIdx.x;   // code 0..1023
  int c   = threadIdx.x;  // channel 0..255
  float e = emb[row * D + c];
  __shared__ float sred[256];
  sred[c] = e * e;
  __syncthreads();
  for (int s = 128; s > 0; s >>= 1) {
    if (c < s) sred[c] += sred[c + s];
    __syncthreads();
  }
  if (c == 0) enorm[row] = sred[0];
  // B fragment layout (16-bit 32x16): lane = half*16 + n, element i<8 -> k = half*8 + i, i>=8 -> k = 16 + half*8 + (i-8)
  int cb = row >> 4, n = row & 15;
  int kc = c >> 5, r = c & 31;
  int half = (r >> 3) & 1, i = (r & 7) | ((r >> 4) << 3);
  int lane = half * 16 + n;
  embb[(((cb * 8 + kc) * 32 + lane) << 4) + i] = (__bf16)e;
}

// ---- Kernel 2: WMMA distance scores + fused argmin (double-buffered B) ----
__global__ void __launch_bounds__(256) vq_argmin(const float* __restrict__ z,
                                                 const float* __restrict__ enorm,
                                                 const __bf16* __restrict__ embb,
                                                 int* __restrict__ idx) {
  int lane = threadIdx.x & 31;
  int wave = threadIdx.x >> 5;
  int tile = blockIdx.x * 8 + wave;   // 4096 tiles of 16 pixels
  int half = lane >> 4;
  int n16  = lane & 15;
  int pix  = tile * 16 + n16;
  size_t zb = (size_t)(pix >> 12) * CHW + (size_t)(pix & (HW - 1));

  // A fragments (loaded once, reused over all 64 code blocks), documented
  // 16-bit A layout: element i<8 -> k = k0+i ; i>=8 -> k = k0+16+(i-8),
  // k0 = kc*32 + half*8.
  v16bf A[8];
#pragma unroll
  for (int kc = 0; kc < 8; ++kc) {
    int k0 = kc * 32 + half * 8;
#pragma unroll
    for (int i = 0; i < 8; ++i) {
      A[kc][i]     = (__bf16)z[zb + (size_t)(k0 + i) * HW];
      A[kc][i + 8] = (__bf16)z[zb + (size_t)(k0 + 16 + i) * HW];
    }
  }

  const v16bf* Bp = (const v16bf*)embb + lane;
  float bestv[8];
  int   besti[8];
#pragma unroll
  for (int v = 0; v < 8; ++v) { bestv[v] = 3.4e38f; besti[v] = 0; }

  // One pipeline step: prefetch B for cb+1 into Bn while computing cb from Bc.
  auto step = [&](int cb, v16bf (&Bc)[8], v16bf (&Bn)[8]) {
    int nxt = (cb + 1) & 63;   // wrap: harmless redundant prefetch on last step
#pragma unroll
    for (int kc = 0; kc < 8; ++kc) Bn[kc] = Bp[(nxt * 8 + kc) * 32];
    v8f acc = {};
#pragma unroll
    for (int kc = 0; kc < 8; ++kc) {
      acc = __builtin_amdgcn_wmma_f32_16x16x32_bf16(
          false, A[kc], false, Bc[kc], (short)0, acc, false, false);
    }
    float en  = enorm[cb * 16 + n16];
    int  code = cb * 16 + n16;
    // C/D layout: acc[v] is row (half*8 + v), column n16 of this code block
#pragma unroll
    for (int v = 0; v < 8; ++v) {
      float d = en - 2.0f * acc[v];   // ||E||^2 - 2 z.E  (||z||^2 const per row)
      if (d < bestv[v]) { bestv[v] = d; besti[v] = code; }
    }
  };

  v16bf B0[8], B1[8];
#pragma unroll
  for (int kc = 0; kc < 8; ++kc) B0[kc] = Bp[kc * 32];
  for (int cb = 0; cb < 64; cb += 2) {
    step(cb,     B0, B1);
    step(cb + 1, B1, B0);
  }

  // min+index reduction across the 16 code-columns (lanes) of each half
#pragma unroll
  for (int m = 8; m >= 1; m >>= 1) {
#pragma unroll
    for (int v = 0; v < 8; ++v) {
      float ov = __shfl_xor(bestv[v], m, 32);
      int   oi = __shfl_xor(besti[v], m, 32);
      if (ov < bestv[v] || (ov == bestv[v] && oi < besti[v])) {
        bestv[v] = ov; besti[v] = oi;
      }
    }
  }
  if (n16 == 0) {
    int rowbase = tile * 16 + half * 8;
#pragma unroll
    for (int v = 0; v < 8; ++v) idx[rowbase + v] = besti[v];
  }
}

// ---- Kernel 3: gather codebook rows, STE output, both tuple outputs ----
__global__ void __launch_bounds__(256) vq_gather(const float4* __restrict__ z4,
                                                 const float* __restrict__ emb,
                                                 const int* __restrict__ idx,
                                                 float4* __restrict__ out4) {
  int t  = blockIdx.x * 256 + threadIdx.x;  // flat over NELEM/4
  int e4 = t & 1023;                        // (h*64+w)/4
  int c  = (t >> 10) & 255;
  int b  = t >> 18;
  int n  = b * HW + (e4 << 2);              // pixel index of first element
  const int* ip = idx + n;
  float4 zi = z4[t];
  float q0 = emb[ip[0] * D + c];
  float q1 = emb[ip[1] * D + c];
  float q2 = emb[ip[2] * D + c];
  float q3 = emb[ip[3] * D + c];
  float4 zq;
  zq.x = zi.x + (q0 - zi.x);
  zq.y = zi.y + (q1 - zi.y);
  zq.z = zi.z + (q2 - zi.z);
  zq.w = zi.w + (q3 - zi.w);
  out4[t] = zq;                               // z_q (straight-through arithmetic)
  float4 q; q.x = q0; q.y = q1; q.z = q2; q.w = q3;
  out4[t + NELEM / 4] = q;                    // quantized
}

extern "C" void kernel_launch(void* const* d_in, const int* in_sizes, int n_in,
                              void* d_out, int out_size, void* d_ws, size_t ws_size,
                              hipStream_t stream) {
  const float* z   = (const float*)d_in[0];
  const float* emb = (const float*)d_in[1];
  float* out = (float*)d_out;

  char* ws = (char*)d_ws;
  float*  enorm = (float*)ws;
  __bf16* embb  = (__bf16*)(ws + 4096);
  int*    idx   = (int*)(ws + 4096 + 524288);

  vq_prep  <<<1024,  256, 0, stream>>>(emb, enorm, embb);
  vq_argmin<<<512,   256, 0, stream>>>(z, enorm, embb, idx);
  vq_gather<<<16384, 256, 0, stream>>>((const float4*)z, emb, idx, (float4*)out);
}
